// Joshi2020_49039936586493
// MI455X (gfx1250) — matrix operand order repaired and checked by reading, not compiled, not run on hardware
//
#include <hip/hip_runtime.h>
#include <hip/hip_bf16.h>
#include <stdint.h>

// ---------------- constants ----------------
constexpr int NTOK   = 1024;
constexpr int SPANW  = 30;
constexpr int NC     = NTOK * SPANW;     // 30720 candidates
constexpr int DMODEL = 768;
constexpr int FEAT   = 20;
constexpr int SPAN_DIM = 2324;
constexpr int KP1    = 2336;             // SPAN_DIM padded to 73*32
constexpr int KT1    = KP1 / 32;         // 73 k-tiles
constexpr int NTOP   = 409;              // int(0.4*1024)
constexpr int TP     = 416;              // NTOP padded to 26*16 (M side)
constexpr int TPB    = 512;              // biaffine N padded to 32 tiles (branch-free WMMA)
constexpr int FF     = 1000;
constexpr int NPAD_FF = 1024;            // FFNN hidden padded to 64 tiles (branch-free WMMA)
constexpr int KANT   = 50;
constexpr int PAIR_DIM = 7052;
constexpr int KP2    = 7168;             // padded to 14*512
constexpr int KS     = 512;              // LDS K-slice for pair GEMM (power of two)
constexpr int NSLICE = KP2 / KS;         // 14
constexpr int NPAIR  = NTOP * KANT;      // 20450
constexpr int MT_PAIR = (NPAIR + 15) / 16; // 1279
constexpr float NEGF = -1e9f;

// ---------------- vector types ----------------
typedef __attribute__((ext_vector_type(16))) _Float16 v16h;
typedef __attribute__((ext_vector_type(8)))  _Float16 v8h;
typedef __attribute__((ext_vector_type(8)))  float    v8f;

__device__ __forceinline__ v8f wmma_f16(v16h a, v16h b, v8f c) {
  // D = A(16x32 f16) * B(32x16 f16) + C(16x16 f32)
  return __builtin_amdgcn_wmma_f32_16x16x32_f16(false, a, false, b, (short)0, c,
                                                false, false);
}

// A fragment, 16x32 f16, per ISA layout:
// lanes 0-15: M=lane, K = {0..7, 16..23}; lanes 16-31: M=lane-16, K = {8..15, 24..31}
__device__ __forceinline__ v16h load_a_frag(const _Float16* A, int lane, int strideK) {
  int m  = lane & 15;
  int kb = (lane < 16) ? 0 : 8;
  const _Float16* p = A + (size_t)m * strideK + kb;
  v8h lo = *(const v8h*)(p);
  v8h hi = *(const v8h*)(p + 16);
  return __builtin_shufflevector(lo, hi, 0,1,2,3,4,5,6,7,8,9,10,11,12,13,14,15);
}

// B fragment, 32x16 f16 stored as N x K row-major (i.e. W^T):
// lanes 0-15: N=lane, K = {0..7, 16..23}; lanes 16-31: N=lane-16, K = {8..15, 24..31}
__device__ __forceinline__ v16h load_b_frag(const _Float16* BT, int lane, int strideK) {
  int nn = lane & 15;
  int kb = (lane < 16) ? 0 : 8;
  const _Float16* p = BT + (size_t)nn * strideK + kb;
  v8h lo = *(const v8h*)(p);
  v8h hi = *(const v8h*)(p + 16);
  return __builtin_shufflevector(lo, hi, 0,1,2,3,4,5,6,7,8,9,10,11,12,13,14,15);
}

__device__ __forceinline__ int bucketDist(int d) {
  if (d <= 4) return d < 0 ? 0 : d;
  int l = 31 - __clz(d) + 3;
  return l > 9 ? 9 : l;
}

// ---------------- prep kernels ----------------
// dst[n*Kp + k] = src[k*N + n] (zero padded)
__global__ void k_transpose_pad_f16(const float* __restrict__ src, _Float16* __restrict__ dst,
                                    int K, int N, int Kp, int Np) {
  long total = (long)Np * Kp;
  long stride = (long)gridDim.x * blockDim.x;
  for (long idx = (long)blockIdx.x * blockDim.x + threadIdx.x; idx < total; idx += stride) {
    int n = (int)(idx / Kp), k = (int)(idx % Kp);
    dst[idx] = (n < N && k < K) ? (_Float16)src[(long)k * N + n] : (_Float16)0.0f;
  }
}

// dst[r*Cp + c] = src[r*Cc + c] (zero padded)
__global__ void k_pad_f16(const float* __restrict__ src, _Float16* __restrict__ dst,
                          int R, int Cc, int Rp, int Cp) {
  long total = (long)Rp * Cp;
  long stride = (long)gridDim.x * blockDim.x;
  for (long idx = (long)blockIdx.x * blockDim.x + threadIdx.x; idx < total; idx += stride) {
    int r = (int)(idx / Cp), c = (int)(idx % Cp);
    dst[idx] = (r < R && c < Cc) ? (_Float16)src[(long)r * Cc + c] : (_Float16)0.0f;
  }
}

__global__ void k_token_attn(const float* __restrict__ tok, const float* __restrict__ w,
                             const float* __restrict__ b, float* __restrict__ out) {
  int i = blockIdx.x * blockDim.x + threadIdx.x;
  if (i >= NTOK) return;
  float s = b[0];
  for (int d = 0; d < DMODEL; ++d) s += tok[(size_t)i * DMODEL + d] * w[d];
  out[i] = s;
}

__global__ void k_width_prior(const float* __restrict__ emb, const float* __restrict__ w1,
                              const float* __restrict__ b1, const float* __restrict__ w2,
                              const float* __restrict__ b2, float* __restrict__ out) {
  __shared__ float red[256];
  int wdt = blockIdx.x, tid = threadIdx.x;
  float s = 0.f;
  for (int h = tid; h < FF; h += 256) {
    float a = b1[h];
    for (int f = 0; f < FEAT; ++f) a += emb[wdt * FEAT + f] * w1[f * FF + h];
    a = a > 0.f ? a : 0.f;
    s += a * w2[h];
  }
  red[tid] = s; __syncthreads();
  for (int st = 128; st > 0; st >>= 1) { if (tid < st) red[tid] += red[tid + st]; __syncthreads(); }
  if (tid == 0) out[wdt] = red[0] + b2[0];
}

__global__ void k_dist_prior(const float* __restrict__ emb, const float* __restrict__ w,
                             const float* __restrict__ b, float* __restrict__ out) {
  int i = threadIdx.x;
  if (i < 10) {
    float s = b[0];
    for (int f = 0; f < FEAT; ++f) s += emb[i * FEAT + f] * w[f];
    out[i] = s;
  }
}

// ---------------- candidate span embeddings ----------------
__global__ void k_cand(const float* __restrict__ tok, const float* __restrict__ tokAttn,
                       const float* __restrict__ embW, _Float16* __restrict__ cand) {
  int c = blockIdx.x;
  int lane = threadIdx.x;  // blockDim 32
  int start = c / SPANW, width = c % SPANW;
  int end = start + width; if (end > NTOK - 1) end = NTOK - 1;
  int L = end - start + 1;
  __shared__ float sp[32];
  float a = (lane < L) ? tokAttn[start + lane] : -3.0e38f;
  float m = a;
  for (int s = 16; s > 0; s >>= 1) { float o = __shfl_xor(m, s, 32); m = o > m ? o : m; }
  float e = (lane < L) ? __expf(a - m) : 0.f;
  float ssum = e;
  for (int s = 16; s > 0; s >>= 1) ssum += __shfl_xor(ssum, s, 32);
  sp[lane] = e / ssum;
  __syncthreads();
  _Float16* out = cand + (size_t)c * KP1;
  for (int i = lane; i < KP1; i += 32) {
    float v;
    if (i < DMODEL) v = tok[(size_t)start * DMODEL + i];
    else if (i < 2 * DMODEL) v = tok[(size_t)end * DMODEL + (i - DMODEL)];
    else if (i < 2 * DMODEL + FEAT) v = embW[width * FEAT + (i - 2 * DMODEL)];
    else if (i < SPAN_DIM) {
      int d = i - (2 * DMODEL + FEAT);
      float s = 0.f;
      for (int t = 0; t < L; ++t) s += sp[t] * tok[(size_t)(start + t) * DMODEL + d];
      v = s;
    } else v = 0.f;
    out[i] = (_Float16)v;
  }
}

// ---------------- mention FFNN (WMMA, fused layer2, branch-free N=64 tiles) ----------------
__global__ __launch_bounds__(256) void k_gemm_ment(
    const _Float16* __restrict__ cand, const _Float16* __restrict__ w1T,
    const float* __restrict__ b1, const float* __restrict__ w2, const float* __restrict__ b2,
    const float* __restrict__ widthPrior, const int* __restrict__ sentMap,
    float* __restrict__ ment) {
  __shared__ float shs[16];
  const int tid = threadIdx.x, lane = tid & 31;
  const int wv = __builtin_amdgcn_readfirstlane(tid >> 5);
  const int m0 = blockIdx.x * 16;
  if (tid < 16) shs[tid] = 0.f;
  __syncthreads();

  v8f zero = {0.f,0.f,0.f,0.f,0.f,0.f,0.f,0.f};
  v8f acc[8];
#pragma unroll
  for (int j = 0; j < 8; ++j) acc[j] = zero;

  const _Float16* Abase = cand + (size_t)m0 * KP1;
  const _Float16* Bbase = w1T + (size_t)(wv * 16) * KP1;
  for (int kt = 0; kt < KT1; ++kt) {
    if (kt + 4 < KT1)
      __builtin_prefetch(Abase + (size_t)(lane & 15) * KP1 + (kt + 4) * 32, 0, 3);
    v16h af = load_a_frag(Abase + kt * 32, lane, KP1);
    v16h bf[8];
#pragma unroll
    for (int j = 0; j < 8; ++j)
      bf[j] = load_b_frag(Bbase + (size_t)(8 * j * 16) * KP1 + kt * 32, lane, KP1);
#pragma unroll
    for (int j = 0; j < 8; ++j)
      acc[j] = wmma_f16(af, bf[j], acc[j]);
  }
  float sacc[8] = {0.f,0.f,0.f,0.f,0.f,0.f,0.f,0.f};
#pragma unroll
  for (int j = 0; j < 8; ++j) {
    int nt = wv + 8 * j;
    int col = nt * 16 + (lane & 15);
    float b1v = (col < FF) ? b1[col] : 0.f;
    float w2v = (col < FF) ? w2[col] : 0.f;
#pragma unroll
    for (int v = 0; v < 8; ++v) {
      float h = acc[j][v] + b1v;
      h = h > 0.f ? h : 0.f;
      sacc[v] += h * w2v;
    }
  }
  const int rbase = (lane >> 4) * 8;
#pragma unroll
  for (int v = 0; v < 8; ++v) atomicAdd(&shs[rbase + v], sacc[v]);
  __syncthreads();
  if (tid < 16) {
    int c = m0 + tid;
    int start = c / SPANW, width = c % SPANW;
    int end = start + width;
    int endc = end < (NTOK - 1) ? end : (NTOK - 1);
    bool valid = (end < NTOK) && (sentMap[start] == sentMap[endc]);
    ment[c] = valid ? (shs[tid] + b2[0] + widthPrior[width]) : NEGF;
  }
}

// ---------------- top-409 spans ----------------
__global__ void k_topk_spans(const float* __restrict__ ment, int* __restrict__ topIdx) {
  extern __shared__ char smem[];
  float* sv  = (float*)smem;                                   // NC floats
  float* rbv = (float*)(smem + (size_t)NC * 4);                // 1024
  int*   rbi = (int*)(smem + (size_t)NC * 4 + 1024 * 4);       // 1024
  unsigned char* sel = (unsigned char*)(smem + (size_t)NC * 4 + 1024 * 8); // NC bytes
  int tid = threadIdx.x;  // blockDim 1024
  for (int i = tid; i < NC; i += 1024) { sv[i] = ment[i]; sel[i] = 0; }
  __syncthreads();
  for (int it = 0; it < NTOP; ++it) {
    float bv = -3.0e38f; int bi = 0;
    for (int i = tid; i < NC; i += 1024) { float v = sv[i]; if (v > bv) { bv = v; bi = i; } }
    rbv[tid] = bv; rbi[tid] = bi;
    __syncthreads();
    for (int s = 512; s > 0; s >>= 1) {
      if (tid < s && rbv[tid + s] > rbv[tid]) { rbv[tid] = rbv[tid + s]; rbi[tid] = rbi[tid + s]; }
      __syncthreads();
    }
    if (tid == 0) { int j = rbi[0]; sel[j] = 1; sv[j] = -3.0e38f; }
    __syncthreads();
  }
  // compact selected indices in ascending order
  int cnt = 0;
  for (int i = tid * SPANW; i < tid * SPANW + SPANW; ++i) cnt += sel[i];
  rbi[tid] = cnt;
  __syncthreads();
  if (tid == 0) { int acc = 0; for (int t = 0; t < 1024; ++t) { int c = rbi[t]; rbi[t] = acc; acc += c; } }
  __syncthreads();
  int off = rbi[tid];
  for (int i = tid * SPANW; i < tid * SPANW + SPANW; ++i)
    if (sel[i]) topIdx[off++] = i;
  if (tid < TP - NTOP) topIdx[NTOP + tid] = 0;
}

// ---------------- gather top spans (TPB blocks: topE 416 rows, ybar 512 rows) ------------
__global__ void k_gather_top(const _Float16* __restrict__ cand, const float* __restrict__ ment,
                             const int* __restrict__ topIdx, const int* __restrict__ spkIds,
                             const int* __restrict__ segIds,
                             _Float16* __restrict__ topE, _Float16* __restrict__ ybar,
                             int* __restrict__ topStarts, int* __restrict__ topEnds,
                             float* __restrict__ topMent, int* __restrict__ topSpk,
                             int* __restrict__ topSeg, float* __restrict__ outStarts,
                             float* __restrict__ outEnds) {
  int x = blockIdx.x, tid = threadIdx.x;  // grid = TPB
  bool real = (x < NTOP);
  int ti = real ? topIdx[x] : 0;
  const _Float16* src = cand + (size_t)ti * KP1;
  for (int k = tid; k < KP1; k += blockDim.x) {
    _Float16 cv = real ? src[k] : (_Float16)0.0f;
    if (x < TP) topE[(size_t)x * KP1 + k] = cv;
    _Float16 yv = cv;
    if (k == SPAN_DIM) yv = real ? (_Float16)1.0f : (_Float16)0.0f; // bias column
    ybar[(size_t)x * KP1 + k] = yv;
  }
  if (tid == 0 && x < TP) {
    int start = ti / SPANW, width = ti % SPANW;
    int end = start + width; if (end > NTOK - 1) end = NTOK - 1;
    if (!real) { start = 0; end = 0; }
    topStarts[x] = start; topEnds[x] = end;
    topMent[x] = real ? ment[ti] : 0.f;
    topSpk[x] = spkIds[start]; topSeg[x] = segIds[start];
    if (real) { outStarts[x] = (float)start; outEnds[x] = (float)end; }
  }
}

// ---------------- biaffine GEMM 1: t2B[y,i] = sum_j W[i,j]*ybar[y,j] (N=32 tiles) --------
__global__ __launch_bounds__(256) void k_gemm_biaff1(
    const _Float16* __restrict__ Wb, const _Float16* __restrict__ ybar,
    _Float16* __restrict__ t2B) {
  const int tid = threadIdx.x, lane = tid & 31;
  const int wv = __builtin_amdgcn_readfirstlane(tid >> 5);
  const int m0 = blockIdx.x * 16;   // rows of W (i)
  v8f zero = {0.f,0.f,0.f,0.f,0.f,0.f,0.f,0.f};
  v8f acc[4];
#pragma unroll
  for (int j = 0; j < 4; ++j) acc[j] = zero;
  const _Float16* Abase = Wb + (size_t)m0 * KP1;
  const _Float16* Bbase = ybar + (size_t)(wv * 16) * KP1;
  for (int kt = 0; kt < KT1; ++kt) {
    v16h af = load_a_frag(Abase + kt * 32, lane, KP1);
    v16h bf[4];
#pragma unroll
    for (int j = 0; j < 4; ++j)
      bf[j] = load_b_frag(Bbase + (size_t)(8 * j * 16) * KP1 + kt * 32, lane, KP1);
#pragma unroll
    for (int j = 0; j < 4; ++j)
      acc[j] = wmma_f16(af, bf[j], acc[j]);
  }
#pragma unroll
  for (int j = 0; j < 4; ++j) {
    int nt = wv + 8 * j;
    int y = nt * 16 + (lane & 15);
    int ib = m0 + (lane >> 4) * 8;
#pragma unroll
    for (int v = 0; v < 8; ++v)
      t2B[(size_t)y * KP1 + (ib + v)] = (_Float16)acc[j][v];  // store transposed (N x K)
  }
}

// ---------------- biaffine GEMM 2 + coarse epilogue ----------------
__global__ __launch_bounds__(256) void k_gemm_biaff2(
    const _Float16* __restrict__ topE, const _Float16* __restrict__ t2B,
    const float* __restrict__ topMent, const float* __restrict__ distPrior,
    float* __restrict__ coarse) {
  const int tid = threadIdx.x, lane = tid & 31;
  const int wv = __builtin_amdgcn_readfirstlane(tid >> 5);
  const int m0 = blockIdx.x * 16;   // rows x
  v8f zero = {0.f,0.f,0.f,0.f,0.f,0.f,0.f,0.f};
  v8f acc[4];
#pragma unroll
  for (int j = 0; j < 4; ++j) acc[j] = zero;
  const _Float16* Abase = topE + (size_t)m0 * KP1;
  const _Float16* Bbase = t2B + (size_t)(wv * 16) * KP1;
  for (int kt = 0; kt < KT1; ++kt) {
    v16h af = load_a_frag(Abase + kt * 32, lane, KP1);
    v16h bf[4];
#pragma unroll
    for (int j = 0; j < 4; ++j)
      bf[j] = load_b_frag(Bbase + (size_t)(8 * j * 16) * KP1 + kt * 32, lane, KP1);
#pragma unroll
    for (int j = 0; j < 4; ++j)
      acc[j] = wmma_f16(af, bf[j], acc[j]);
  }
#pragma unroll
  for (int j = 0; j < 4; ++j) {
    int nt = wv + 8 * j;
    int y = nt * 16 + (lane & 15);
    int xb = m0 + (lane >> 4) * 8;
#pragma unroll
    for (int v = 0; v < 8; ++v) {
      int x = xb + v;
      if (x < NTOP && y < NTOP) {
        int offd = x - y;
        float c = acc[j][v] + topMent[x] + topMent[y];
        c += (offd >= 1) ? 0.f : NEGF;
        int od = offd > 0 ? offd : 0;
        c += distPrior[bucketDist(od)];
        coarse[(size_t)x * TP + y] = c;
      }
    }
  }
}

// ---------------- top-50 antecedents per span ----------------
__global__ void k_topk_ant(const float* __restrict__ coarse, int* __restrict__ antIdx,
                           float* __restrict__ antCoarse, int* __restrict__ antOff,
                           float* __restrict__ outAntIdx) {
  __shared__ float row[TP];
  __shared__ float rv[128];
  __shared__ int ri[128];
  int x = blockIdx.x, tid = threadIdx.x;  // blockDim 128
  for (int i = tid; i < TP; i += 128) row[i] = (i < NTOP) ? coarse[(size_t)x * TP + i] : -3.0e38f;
  __syncthreads();
  for (int it = 0; it < KANT; ++it) {
    float bv = -3.3e38f; int bi = 0;
    for (int i = tid; i < NTOP; i += 128) { if (row[i] > bv) { bv = row[i]; bi = i; } }
    rv[tid] = bv; ri[tid] = bi;
    __syncthreads();
    for (int s = 64; s > 0; s >>= 1) {
      if (tid < s && rv[tid + s] > rv[tid]) { rv[tid] = rv[tid + s]; ri[tid] = ri[tid + s]; }
      __syncthreads();
    }
    if (tid == 0) {
      int j = ri[0];
      antIdx[x * KANT + it] = j;
      antCoarse[x * KANT + it] = rv[0];
      antOff[x * KANT + it] = x - j;
      outAntIdx[x * KANT + it] = (float)j;
      row[j] = -3.3e38f;
    }
    __syncthreads();
  }
}

__global__ void k_init_final(float* __restrict__ finalOut) {
  int i = blockIdx.x * blockDim.x + threadIdx.x;
  if (i < NTOP) finalOut[(size_t)i * (KANT + 1)] = 0.f;
}

// ---------------- pair FFNN (LDS-staged A, WMMA, fused layer2 + final) ----------------
__global__ __launch_bounds__(256) void k_gemm_pair(
    const _Float16* __restrict__ topE, const _Float16* __restrict__ wp1T,
    const float* __restrict__ b1, const float* __restrict__ w2, const float* __restrict__ b2,
    const int* __restrict__ antIdx, const int* __restrict__ antOff,
    const float* __restrict__ antCoarse,
    const int* __restrict__ topSpk, const int* __restrict__ topSeg,
    const int* __restrict__ genre_p,
    const float* __restrict__ embSpk, const float* __restrict__ embGenre,
    const float* __restrict__ embSeg, const float* __restrict__ embAntDist,
    float* __restrict__ finalOut) {
  __shared__ __align__(32) _Float16 As[16 * KS];
  __shared__ __align__(32) _Float16 sfeat[16 * 80];
  __shared__ int sx[16], sa[16], soff[16];
  __shared__ float shs[16];
  const int tid = threadIdx.x, lane = tid & 31;
  const int wv = __builtin_amdgcn_readfirstlane(tid >> 5);
  const int m0 = blockIdx.x * 16;
  const int genre = genre_p[0];
  // per-row metadata
  if (tid < 16) {
    shs[tid] = 0.f;
    int p = m0 + tid;
    int x = 0, a = 0, offv = 0;
    if (p < NPAIR) {
      x = p / KANT;
      int y = p % KANT;
      a = antIdx[x * KANT + y];
      offv = antOff[x * KANT + y];
    }
    sx[tid] = x; sa[tid] = a; soff[tid] = offv;
  }
  __syncthreads();
  // precompute 80 pair-feature dims per row
  for (int idx = tid; idx < 16 * 80; idx += 256) {
    int r = idx / 80, fi = idx % 80;
    int g = fi / FEAT, f = fi % FEAT;
    int x = sx[r], a = sa[r];
    float v;
    if (g == 0) {
      int ss = (topSpk[x] == topSpk[a]) ? 1 : 0;
      v = embSpk[ss * FEAT + f];
    } else if (g == 1) {
      v = embGenre[genre * FEAT + f];
    } else if (g == 2) {
      int sd = topSeg[x] - topSeg[a];
      sd = sd < 0 ? 0 : (sd > 10 ? 10 : sd);
      v = embSeg[sd * FEAT + f];
    } else {
      int od = soff[r]; od = od > 0 ? od : 0;
      v = embAntDist[bucketDist(od) * FEAT + f];
    }
    sfeat[idx] = (_Float16)v;
  }

  v8f zero = {0.f,0.f,0.f,0.f,0.f,0.f,0.f,0.f};
  v8f acc[8];
#pragma unroll
  for (int j = 0; j < 8; ++j) acc[j] = zero;

  const _Float16* Bbase = wp1T + (size_t)(wv * 16) * KP2;
  for (int sl = 0; sl < NSLICE; ++sl) {
    __syncthreads();
    // build A-tile slice: pair rows [m0, m0+16) x K slice [sl*KS, sl*KS+KS)
    for (int idx = tid; idx < 16 * KS; idx += 256) {
      int r = idx >> 9;            // /KS
      int kk = idx & (KS - 1);
      int k = sl * KS + kk;
      int x = sx[r], a = sa[r];
      _Float16 v;
      if (k < SPAN_DIM) {
        v = topE[(size_t)x * KP1 + k];
      } else if (k < 2 * SPAN_DIM) {
        v = topE[(size_t)a * KP1 + (k - SPAN_DIM)];
      } else if (k < 3 * SPAN_DIM) {
        int d = k - 2 * SPAN_DIM;
        v = (_Float16)((float)topE[(size_t)x * KP1 + d] * (float)topE[(size_t)a * KP1 + d]);
      } else if (k < PAIR_DIM) {
        v = sfeat[r * 80 + (k - 3 * SPAN_DIM)];
      } else {
        v = (_Float16)0.0f;
      }
      As[idx] = v;
    }
    __syncthreads();
    for (int kt = 0; kt < KS / 32; ++kt) {
      if (kt + 4 < KS / 32)
        __builtin_prefetch(Bbase + sl * KS + (kt + 4) * 32 + (size_t)(lane & 15) * KP2, 0, 3);
      v16h af = load_a_frag(As + kt * 32, lane, KS);
      v16h bf[8];
#pragma unroll
      for (int j = 0; j < 8; ++j)
        bf[j] = load_b_frag(Bbase + (size_t)(8 * j * 16) * KP2 + sl * KS + kt * 32, lane, KP2);
#pragma unroll
      for (int j = 0; j < 8; ++j)
        acc[j] = wmma_f16(af, bf[j], acc[j]);
    }
  }
  // fused bias + ReLU + w2 dot
  float sacc[8] = {0.f,0.f,0.f,0.f,0.f,0.f,0.f,0.f};
#pragma unroll
  for (int j = 0; j < 8; ++j) {
    int nt = wv + 8 * j;
    int col = nt * 16 + (lane & 15);
    float b1v = (col < FF) ? b1[col] : 0.f;
    float w2v = (col < FF) ? w2[col] : 0.f;
#pragma unroll
    for (int v = 0; v < 8; ++v) {
      float h = acc[j][v] + b1v;
      h = h > 0.f ? h : 0.f;
      sacc[v] += h * w2v;
    }
  }
  const int rbase = (lane >> 4) * 8;
#pragma unroll
  for (int v = 0; v < 8; ++v) atomicAdd(&shs[rbase + v], sacc[v]);
  __syncthreads();
  if (tid < 16) {
    int p = m0 + tid;
    if (p < NPAIR) {
      int x = p / KANT, y = p % KANT;
      float slow = shs[tid] + b2[0];
      bool mk = antOff[x * KANT + y] >= 1;
      finalOut[(size_t)x * (KANT + 1) + 1 + y] = mk ? (slow + antCoarse[x * KANT + y]) : NEGF;
    }
  }
}

// ---------------- host launcher ----------------
extern "C" void kernel_launch(void* const* d_in, const int* in_sizes, int n_in,
                              void* d_out, int out_size, void* d_ws, size_t ws_size,
                              hipStream_t stream) {
  (void)in_sizes; (void)n_in; (void)out_size; (void)ws_size;
  const float* tok           = (const float*)d_in[0];
  const int*   spk           = (const int*)d_in[1];
  const int*   sentMap       = (const int*)d_in[2];
  const int*   segIds        = (const int*)d_in[3];
  const int*   genre         = (const int*)d_in[4];
  const float* embSpanW      = (const float*)d_in[5];
  const float* embSpanWPrior = (const float*)d_in[6];
  const float* wMentAttn     = (const float*)d_in[7];
  const float* bMentAttn     = (const float*)d_in[8];
  const float* wSpan1        = (const float*)d_in[9];
  const float* bSpan1        = (const float*)d_in[10];
  const float* wSpan2        = (const float*)d_in[11];
  const float* bSpan2        = (const float*)d_in[12];
  const float* wWidth1       = (const float*)d_in[13];
  const float* bWidth1       = (const float*)d_in[14];
  const float* wWidth2       = (const float*)d_in[15];
  const float* bWidth2       = (const float*)d_in[16];
  const float* wBiaff        = (const float*)d_in[17];
  const float* embAntDistPr  = (const float*)d_in[18];
  const float* wAntDist      = (const float*)d_in[19];
  const float* bAntDist      = (const float*)d_in[20];
  const float* embSameSpk    = (const float*)d_in[21];
  const float* embGenreT     = (const float*)d_in[22];
  const float* embSegDist    = (const float*)d_in[23];
  const float* embTopAntDist = (const float*)d_in[24];
  const float* wPair1        = (const float*)d_in[25];
  const float* bPair1        = (const float*)d_in[26];
  const float* wPair2        = (const float*)d_in[27];
  const float* bPair2        = (const float*)d_in[28];

  char* ws = (char*)d_ws;
  size_t off = 0;
  auto alloc = [&](size_t bytes) -> char* {
    off = (off + 255) & ~(size_t)255;
    char* p = ws + off;
    off += bytes;
    return p;
  };
  float*    tokAttn    = (float*)alloc((size_t)NTOK * 4);
  float*    widthPrior = (float*)alloc(SPANW * 4);
  float*    distPrior  = (float*)alloc(10 * 4);
  _Float16* cand       = (_Float16*)alloc((size_t)NC * KP1 * 2);
  float*    ment       = (float*)alloc((size_t)NC * 4);
  _Float16* w1T        = (_Float16*)alloc((size_t)NPAD_FF * KP1 * 2);
  _Float16* wp1T       = (_Float16*)alloc((size_t)NPAD_FF * KP2 * 2);
  _Float16* Wb         = (_Float16*)alloc((size_t)KP1 * KP1 * 2);
  int*      topIdx     = (int*)alloc(TP * 4);
  int*      topStarts  = (int*)alloc(TP * 4);
  int*      topEnds    = (int*)alloc(TP * 4);
  float*    topMent    = (float*)alloc(TP * 4);
  int*      topSpk     = (int*)alloc(TP * 4);
  int*      topSeg     = (int*)alloc(TP * 4);
  _Float16* topE       = (_Float16*)alloc((size_t)TP * KP1 * 2);
  _Float16* ybar       = (_Float16*)alloc((size_t)TPB * KP1 * 2);
  _Float16* t2B        = (_Float16*)alloc((size_t)TPB * KP1 * 2);
  float*    coarse     = (float*)alloc((size_t)NTOP * TP * 4);
  int*      antIdx     = (int*)alloc((size_t)NPAIR * 4);
  int*      antOff     = (int*)alloc((size_t)NPAIR * 4);
  float*    antCoarse  = (float*)alloc((size_t)NPAIR * 4);

  float* outStarts = (float*)d_out;
  float* outEnds   = outStarts + NTOP;
  float* outAntIdx = outEnds + NTOP;
  float* outFinal  = outAntIdx + NPAIR;

  k_transpose_pad_f16<<<2048, 256, 0, stream>>>(wSpan1, w1T, SPAN_DIM, FF, KP1, NPAD_FF);
  k_transpose_pad_f16<<<4096, 256, 0, stream>>>(wPair1, wp1T, PAIR_DIM, FF, KP2, NPAD_FF);
  k_pad_f16<<<4096, 256, 0, stream>>>(wBiaff, Wb, SPAN_DIM, SPAN_DIM + 1, KP1, KP1);
  k_token_attn<<<4, 256, 0, stream>>>(tok, wMentAttn, bMentAttn, tokAttn);
  k_width_prior<<<SPANW, 256, 0, stream>>>(embSpanWPrior, wWidth1, bWidth1, wWidth2, bWidth2, widthPrior);
  k_dist_prior<<<1, 32, 0, stream>>>(embAntDistPr, wAntDist, bAntDist, distPrior);
  k_cand<<<NC, 32, 0, stream>>>(tok, tokAttn, embSpanW, cand);
  k_gemm_ment<<<NC / 16, 256, 0, stream>>>(cand, w1T, bSpan1, wSpan2, bSpan2, widthPrior, sentMap, ment);
  size_t shTopk = (size_t)NC * 4 + 1024 * 4 + 1024 * 4 + NC;
  k_topk_spans<<<1, 1024, shTopk, stream>>>(ment, topIdx);
  k_gather_top<<<TPB, 256, 0, stream>>>(cand, ment, topIdx, spk, segIds, topE, ybar,
                                        topStarts, topEnds, topMent, topSpk, topSeg,
                                        outStarts, outEnds);
  k_gemm_biaff1<<<KP1 / 16, 256, 0, stream>>>(Wb, ybar, t2B);
  k_gemm_biaff2<<<TP / 16, 256, 0, stream>>>(topE, t2B, topMent, distPrior, coarse);
  k_topk_ant<<<NTOP, 128, 0, stream>>>(coarse, antIdx, antCoarse, antOff, outAntIdx);
  k_init_final<<<2, 256, 0, stream>>>(outFinal);
  k_gemm_pair<<<MT_PAIR, 256, 0, stream>>>(topE, wp1T, bPair1, wPair2, bPair2, antIdx, antOff,
                                           antCoarse, topSpk, topSeg, genre, embSameSpk,
                                           embGenreT, embSegDist, embTopAntDist, outFinal);
}
